// LearnableLaplacianBasis_60876866454320
// MI455X (gfx1250) — matrix operand order repaired and checked by reading, not compiled, not run on hardware
//
#include <hip/hip_runtime.h>
#include <hip/hip_bf16.h>

typedef __attribute__((ext_vector_type(16))) _Float16 v16h;
typedef __attribute__((ext_vector_type(8)))  float    v8f;
typedef __attribute__((ext_vector_type(4)))  float    f32x4;

#define BB 16
#define TT 4096
#define DD 512
#define KK 256
#define NS 32          // time segments for parallel scan
#define SEGL (TT / NS) // 128 steps per segment

// params layout: [0..K) rho | [K..2K) ct | [2K..3K) st |
//                [3K..4K) R^L cos | [4K..5K) R^L sin | [5K] 1/sigma
#define P_RHO  0
#define P_CT   (KK)
#define P_ST   (2 * KK)
#define P_RLC  (3 * KK)
#define P_RLS  (4 * KK)
#define P_ISIG (5 * KK)

// ---------------- utility device funcs ----------------

__device__ __forceinline__ unsigned pcg_hash(unsigned v) {
    unsigned state = v * 747796405u + 2891336453u;
    unsigned word  = ((state >> ((state >> 28u) + 4u)) ^ state) * 277803737u;
    return (word >> 22u) ^ word;
}
__device__ __forceinline__ float u01(unsigned h) {
    return (float)(h >> 8) * (1.0f / 16777216.0f) + (0.5f / 16777216.0f);
}
__device__ __forceinline__ float softplusf(float x) {
    return (x > 20.f) ? x : log1pf(expf(x));
}
__device__ __forceinline__ float block_sum(float v, float* red, int tid) {
    red[tid] = v; __syncthreads();
    for (int s = 256; s > 0; s >>= 1) {
        if (tid < s) red[tid] += red[tid + s];
        __syncthreads();
    }
    float r = red[0]; __syncthreads();
    return r;
}

// ---------------- kernel 1: spectral norm + per-k params ----------------
__global__ __launch_bounds__(512) void prep_kernel(
    const float* __restrict__ s_real_raw, const float* __restrict__ s_imag,
    const float* __restrict__ tau_raw, const float* __restrict__ W,
    float* __restrict__ params)
{
    __shared__ float u0[KK];
    __shared__ float vsh[DD];
    __shared__ float red[512];
    const int tid = threadIdx.x;

    // u0 ~ N(0,1) (deterministic hash-based; JAX threefry not reproducible here)
    if (tid < KK) {
        float a1 = u01(pcg_hash(2u * tid + 1u));
        float a2 = u01(pcg_hash(2u * tid + 2u));
        u0[tid] = sqrtf(-2.f * logf(a1)) * cosf(6.28318530718f * a2);
    }
    __syncthreads();
    float nrm = sqrtf(block_sum((tid < KK) ? u0[tid] * u0[tid] : 0.f, red, tid));
    float sc = 1.f / fmaxf(nrm, 1e-7f);
    if (tid < KK) u0[tid] *= sc;
    __syncthreads();

    // v = W^T u0  (coalesced: thread tid handles column d = tid)
    {
        float acc = 0.f;
        for (int k = 0; k < KK; ++k) acc += W[(size_t)k * DD + tid] * u0[k];
        vsh[tid] = acc;
    }
    __syncthreads();
    nrm = sqrtf(block_sum(vsh[tid] * vsh[tid], red, tid));
    sc = 1.f / fmaxf(nrm, 1e-7f);
    vsh[tid] *= sc;
    __syncthreads();

    // sigma = u . (W v) with u = Wv/max(||Wv||,eps)  =>  ||Wv||^2 / max(||Wv||,eps)
    float wv = 0.f;
    if (tid < KK) {
        const float* wrow = W + (size_t)tid * DD;
        for (int d = 0; d < DD; ++d) wv += wrow[d] * vsh[d];
    }
    float ssq = block_sum((tid < KK) ? wv * wv : 0.f, red, tid);
    float wvn = sqrtf(ssq);
    float sigma = ssq / fmaxf(wvn, 1e-7f);

    if (tid < KK) {
        float tau    = softplusf(tau_raw[0]) + 1e-3f;
        float alpha0 = (softplusf(s_real_raw[tid]) + 1e-6f) * tau;
        float omega0 = s_imag[tid] * tau;
        const float dt = 1.0f / (float)(TT - 1);
        params[P_RHO + tid] = expf(-alpha0 * dt);
        params[P_CT  + tid] = cosf(omega0 * dt);
        params[P_ST  + tid] = sinf(omega0 * dt);
        // R^L for segment combine: R = rho*rot(theta) => R^L = rho^L * rot(L*theta)
        float rhoL = expf(-alpha0 * dt * (float)SEGL);
        float thL  = omega0 * dt * (float)SEGL;
        params[P_RLC + tid] = rhoL * cosf(thL);
        params[P_RLS + tid] = rhoL * sinf(thL);
    }
    if (tid == 0) params[P_ISIG] = 1.f / fmaxf(sigma, 1e-30f);
}

// ---------------- kernel 2: pack W/sigma into WMMA B-lane layout (f16) ----------------
// Per ISA 7.12.2 (B 32x16, f16): lane L holds column N = nT*16 + (L&15),
// K-range (L>>4)*16..+15 (this code's k order, not the ISA's; A matches) -> 32B per lane.
__global__ __launch_bounds__(256) void pack_b_kernel(
    const float* __restrict__ W, const float* __restrict__ params,
    _Float16* __restrict__ Bp)
{
    int t = blockIdx.x * blockDim.x + threadIdx.x;   // 16*16*32 = 8192 threads
    if (t >= 16 * 16 * 32) return;
    int lane = t & 31;
    int nT   = (t >> 5) & 15;
    int dT   = t >> 9;
    float inv_sigma = params[P_ISIG];
    int n  = nT * 16 + (lane & 15);
    int d0 = dT * 32 + (lane >> 4) * 16;
    const float* wsrc = W + (size_t)n * DD + d0;     // 64B contiguous per thread
    _Float16* dst = Bp + (size_t)t * 16;
    #pragma unroll
    for (int h = 0; h < 16; ++h)
        dst[h] = (_Float16)(wsrc[h] * inv_sigma);
}

// ---------------- kernel 3: WMMA GEMM  U[m,n] = sum_d x[m,d]*Wsn[n,d] + b[n] ----------------
__global__ __launch_bounds__(256) void gemm_kernel(
    const float* __restrict__ x, const _Float16* __restrict__ Bp,
    const float* __restrict__ bias, float* __restrict__ U)
{
    const int wave = threadIdx.x >> 5;
    const int lane = threadIdx.x & 31;
    const int g = lane >> 4;
    const int r = lane & 15;
    const int m0 = (blockIdx.x * 8 + wave) * 16;

    v8f acc[16];
    #pragma unroll
    for (int nT = 0; nT < 16; ++nT) {
        float bv = bias[nT * 16 + r];
        #pragma unroll
        for (int j = 0; j < 8; ++j) acc[nT][j] = bv;
    }

    const float* xrow = x + (size_t)(m0 + r) * DD;
    const v16h* bptr = (const v16h*)Bp;

    for (int d0 = 0; d0 < DD; d0 += 32) {
        f32x4 c0 = *(const f32x4*)(xrow + d0 + g * 16);
        f32x4 c1 = *(const f32x4*)(xrow + d0 + g * 16 + 4);
        f32x4 c2 = *(const f32x4*)(xrow + d0 + g * 16 + 8);
        f32x4 c3 = *(const f32x4*)(xrow + d0 + g * 16 + 12);
        v16h a;
        #pragma unroll
        for (int i = 0; i < 4; ++i) {
            a[i]      = (_Float16)c0[i];
            a[4 + i]  = (_Float16)c1[i];
            a[8 + i]  = (_Float16)c2[i];
            a[12 + i] = (_Float16)c3[i];
        }
        const int dT = d0 >> 5;
        #pragma unroll
        for (int nT = 0; nT < 16; ++nT) {
            v16h bm = bptr[(size_t)(dT * 16 + nT) * 32 + lane];
            acc[nT] = __builtin_amdgcn_wmma_f32_16x16x32_f16(
                false, a, false, bm, (short)0, acc[nT], false, false);
        }
    }

    #pragma unroll
    for (int nT = 0; nT < 16; ++nT) {
        #pragma unroll
        for (int j = 0; j < 8; ++j) {
            int m = m0 + j + 8 * g;
            U[(size_t)m * KK + nT * 16 + r] = acc[nT][j];
        }
    }
}

// ---------------- kernel 4a: per-segment local scan (zero init), end-states only ----
// gid = seg*4096 + b*256 + k  -> lanes cover consecutive k (coalesced 128B loads)
__global__ __launch_bounds__(256) void scan_local_kernel(
    const float* __restrict__ U, const float* __restrict__ params,
    float* __restrict__ carryC, float* __restrict__ carryS)
{
    int gid = blockIdx.x * 256 + threadIdx.x;        // 0 .. NS*BB*KK-1
    int k = gid & (KK - 1);
    int b = (gid >> 8) & (BB - 1);
    int seg = gid >> 12;
    const float rct = params[P_RHO + k] * params[P_CT + k];
    const float rst = params[P_RHO + k] * params[P_ST + k];
    const float* u = U + ((size_t)b * TT + (size_t)seg * SEGL) * KK + k;
    float c = 0.f, s = 0.f;
    #pragma unroll 1
    for (int t = 0; t < SEGL; t += 4) {
        __builtin_prefetch(u + (size_t)(t + 16) * KK, 0, 0);
        #pragma unroll
        for (int q = 0; q < 4; ++q) {
            float ut = u[(size_t)(t + q) * KK];
            float cn = fmaf(rct, c, fmaf(-rst, s, ut));
            float sn = fmaf(rst, c, rct * s);
            c = cn; s = sn;
        }
    }
    carryC[gid] = c;
    carryS[gid] = s;
}

// ---------------- kernel 4b: combine segment carries into entry prefixes ----------
// E_{-1}=0;  pref[seg] = E_{seg-1};  E_seg = Eloc_seg + R^L (.) E_{seg-1}
__global__ __launch_bounds__(256) void combine_kernel(
    const float* __restrict__ carryC, const float* __restrict__ carryS,
    const float* __restrict__ params,
    float* __restrict__ prefC, float* __restrict__ prefS)
{
    int idx = blockIdx.x * 256 + threadIdx.x;        // 0 .. BB*KK-1
    int k = idx & (KK - 1);
    const float RLc = params[P_RLC + k];
    const float RLs = params[P_RLS + k];
    float ec = 0.f, es = 0.f;
    #pragma unroll
    for (int seg = 0; seg < NS; ++seg) {
        size_t off = (size_t)seg * (BB * KK) + idx;
        prefC[off] = ec;
        prefS[off] = es;
        float lc = carryC[off], ls = carryS[off];
        float nc = fmaf(RLc, ec, fmaf(-RLs, es, lc));
        float ns = fmaf(RLs, ec, fmaf(RLc, es, ls));
        ec = nc; es = ns;
    }
}

// ---------------- kernel 4c: seeded per-segment scan, writes output ---------------
__global__ __launch_bounds__(256) void scan_apply_kernel(
    const float* __restrict__ U, const float* __restrict__ params,
    const float* __restrict__ prefC, const float* __restrict__ prefS,
    float* __restrict__ out)
{
    int gid = blockIdx.x * 256 + threadIdx.x;
    int k = gid & (KK - 1);
    int b = (gid >> 8) & (BB - 1);
    int seg = gid >> 12;
    const float rct = params[P_RHO + k] * params[P_CT + k];
    const float rst = params[P_RHO + k] * params[P_ST + k];
    const float* u = U + ((size_t)b * TT + (size_t)seg * SEGL) * KK + k;
    float* o = out + ((size_t)b * TT + (size_t)seg * SEGL) * (2 * KK) + k;
    float c = prefC[gid], s = prefS[gid];
    #pragma unroll 1
    for (int t = 0; t < SEGL; t += 4) {
        __builtin_prefetch(u + (size_t)(t + 16) * KK, 0, 0);
        #pragma unroll
        for (int q = 0; q < 4; ++q) {
            float ut = u[(size_t)(t + q) * KK];
            float cn = fmaf(rct, c, fmaf(-rst, s, ut));
            float sn = fmaf(rst, c, rct * s);
            c = cn; s = sn;
            o[(size_t)(t + q) * (2 * KK)]      = c;
            o[(size_t)(t + q) * (2 * KK) + KK] = s;
        }
    }
}

// ---------------- launch ----------------
extern "C" void kernel_launch(void* const* d_in, const int* in_sizes, int n_in,
                              void* d_out, int out_size, void* d_ws, size_t ws_size,
                              hipStream_t stream) {
    const float* x          = (const float*)d_in[0];
    const float* s_real_raw = (const float*)d_in[1];
    const float* s_imag     = (const float*)d_in[2];
    const float* tau_raw    = (const float*)d_in[3];
    const float* W          = (const float*)d_in[4];
    const float* b          = (const float*)d_in[5];
    float* out = (float*)d_out;

    char* ws = (char*)d_ws;
    const size_t U_BYTES   = (size_t)BB * TT * KK * sizeof(float);          // 64 MB
    const size_t BP_BYTES  = (size_t)16 * 16 * 32 * 16 * sizeof(_Float16);  // 256 KB
    const size_t PAR_BYTES = 32768;                                          // params pad
    const size_t CAR_BYTES = (size_t)NS * BB * KK * sizeof(float);          // 512 KB each

    size_t off = 0;
    float*    U      = (float*)(ws + off); off += U_BYTES;
    _Float16* Bp     = (_Float16*)(ws + off); off += BP_BYTES;
    float*    params = (float*)(ws + off); off += PAR_BYTES;
    float*    carryC = (float*)(ws + off); off += CAR_BYTES;
    float*    carryS = (float*)(ws + off); off += CAR_BYTES;
    float*    prefC  = (float*)(ws + off); off += CAR_BYTES;
    float*    prefS  = (float*)(ws + off); off += CAR_BYTES;

    prep_kernel      <<<1,   512, 0, stream>>>(s_real_raw, s_imag, tau_raw, W, params);
    pack_b_kernel    <<<32,  256, 0, stream>>>(W, params, Bp);
    gemm_kernel      <<<512, 256, 0, stream>>>(x, Bp, b, U);
    scan_local_kernel<<<(NS * BB * KK) / 256, 256, 0, stream>>>(U, params, carryC, carryS);
    combine_kernel   <<<(BB * KK) / 256,      256, 0, stream>>>(carryC, carryS, params, prefC, prefS);
    scan_apply_kernel<<<(NS * BB * KK) / 256, 256, 0, stream>>>(U, params, prefC, prefS, out);
}
